// Discriminator_87660282511381
// MI455X (gfx1250) — matrix operand order repaired and checked by reading, not compiled, not run on hardware
//
#include <hip/hip_runtime.h>
#include <math.h>

typedef __attribute__((ext_vector_type(16))) _Float16 v16h;
typedef __attribute__((ext_vector_type(8)))  float    v8f;
typedef __attribute__((ext_vector_type(4)))  unsigned int u32x4;
typedef __attribute__((ext_vector_type(8)))  int      i32x8;
typedef __attribute__((ext_vector_type(4)))  int      i32x4;

#define B_   512
#define L_   50
#define E_   512
#define C1   20
#define C2   40
#define HP   25     // pooled H/W after outer-sum pool
#define HC   23     // conv2 output H/W
#define HO   11     // final pooled H/W
#define K2   180    // conv2 reduction (20*3*3)
#define K2P  192    // padded to multiple of 32
#define NPIX 529    // 23*23
#define MT   34     // ceil(529/16)
#define PLZ  13104  // 20*625 LDS plane + zero pad region

__device__ __forceinline__ v8f wmma_f16(v16h a, v16h b, v8f c) {
  // D = A(16x32,f16) * B(32x16,f16) + C(16x16,f32)
  return __builtin_amdgcn_wmma_f32_16x16x32_f16(false, a, false, b, (short)0, c, false, false);
}

// ---------------------------------------------------------------------------
// K0: one-time weight prep — pad + convert W1 (20x512 -> 32x512) and
// W2 (40x180 -> 48x192) to f16 in workspace. Removes all per-block converts
// and the divergent zero-fill path seen in round-0 codegen.
// ---------------------------------------------------------------------------
__global__ void k0_prep(const float* __restrict__ W1, const float* __restrict__ W2,
                        _Float16* __restrict__ w1h, _Float16* __restrict__ w2h) {
  const int stride = gridDim.x * blockDim.x;
  int tid = blockIdx.x * blockDim.x + threadIdx.x;
  for (int idx = tid; idx < 32 * E_; idx += stride) {
    int c = idx / E_, k = idx % E_;
    w1h[idx] = (_Float16)((c < C1) ? W1[c * E_ + k] : 0.f);
  }
  for (int idx = tid; idx < 48 * K2P; idx += stride) {
    int n = idx / K2P, k = idx % K2P;
    w2h[idx] = (_Float16)((n < C2 && k < K2) ? W2[n * K2 + k] : 0.f);
  }
}

// ---------------------------------------------------------------------------
// K1: embedding gather + GEMM  f[row,c] = sum_e emb[tok[row],e] * W1[c,e] + b1[c]
// One wave per 16-row M tile; N padded 20 -> 32 (two 16-wide tiles).
// B fragments are read straight from the prepadded f16 weights in global
// (one global_load_b128 per fragment half; L2-resident, shared by all blocks).
// ---------------------------------------------------------------------------
__global__ void __launch_bounds__(32)
k1_embed_gemm(const int* __restrict__ tok, const float* __restrict__ emb,
              const _Float16* __restrict__ w1h, const float* __restrict__ bias1,
              float* __restrict__ fout) {
  __shared__ _Float16 As[16 * 32];   // [row][k]
  __shared__ int rtok[16];
  const int lane = threadIdx.x;
  const int row0 = blockIdx.x * 16;
  if (lane < 16) rtok[lane] = tok[row0 + lane];
  __syncthreads();
  const int r  = lane >> 1;          // row loaded by this lane
  const int hs = (lane & 1) * 16;    // half-row select
  const int m  = lane & 15;          // fragment row / col lane index
  const int h  = lane >> 4;          // fragment half select
  v8f acc0 = {}, acc1 = {};
  for (int k0 = 0; k0 < E_; k0 += 32) {
    const float* src = emb + (size_t)rtok[r] * E_ + k0 + hs;
    __builtin_prefetch(src + 32, 0, 0);
#pragma unroll
    for (int i = 0; i < 16; ++i) As[r * 32 + hs + i] = (_Float16)src[i];
    __syncthreads();
    v16h a, b0, b1;
#pragma unroll
    for (int e = 0; e < 8; ++e) a[e] = As[m * 32 + 16 * h + e];          // K = 16h + e
#pragma unroll
    for (int e = 0; e < 8; ++e) a[8 + e] = As[m * 32 + 16 * h + 8 + e]; // K = 16h + 8 + e
#pragma unroll
    for (int e = 0; e < 16; ++e) b0[e] = w1h[m * E_ + k0 + 16 * h + e];        // N tile 0
#pragma unroll
    for (int e = 0; e < 16; ++e) b1[e] = w1h[(16 + m) * E_ + k0 + 16 * h + e]; // N tile 1
    acc0 = wmma_f16(a, b0, acc0);
    acc1 = wmma_f16(a, b1, acc1);
    __syncthreads();
  }
#pragma unroll
  for (int rr = 0; rr < 8; ++rr) {
    int row = row0 + rr + 8 * h;
    fout[row * C1 + m] = acc0[rr] + bias1[m];
    if (m + 16 < C1) fout[row * C1 + 16 + m] = acc1[rr] + bias1[16 + m];
  }
}

// ---------------------------------------------------------------------------
// K2: per-sample BatchNorm1d (stats over L) + ReLU, in place
// ---------------------------------------------------------------------------
__global__ void k2_bn1(float* __restrict__ f, const float* __restrict__ g,
                       const float* __restrict__ beta) {
  const int b = blockIdx.x, c = threadIdx.x;
  if (c >= C1) return;
  float sum = 0.f, ss = 0.f;
  for (int l = 0; l < L_; ++l) { float v = f[(b * L_ + l) * C1 + c]; sum += v; ss += v * v; }
  const float mu  = sum * (1.f / L_);
  const float var = ss * (1.f / L_) - mu * mu;
  const float sc  = g[c] * rsqrtf(var + 1e-5f);
  const float sh  = beta[c] - mu * sc;
  for (int l = 0; l < L_; ++l) {
    int i = (b * L_ + l) * C1 + c;
    f[i] = fmaxf(0.f, f[i] * sc + sh);
  }
}

// ---------------------------------------------------------------------------
// K3: pairwise maxes along L (2x2 maxpool distributes over the outer sum, so
// the [B,50,50,20] tensor is never materialized)
// ---------------------------------------------------------------------------
__global__ void k3_poolmax(const float* __restrict__ s, const float* __restrict__ t,
                           float* __restrict__ smx, float* __restrict__ tmx) {
  const int b = blockIdx.x, tid = threadIdx.x;
  if (tid >= HP * C1) return;
  const int p = tid / C1, c = tid % C1;
  float a0 = s[(b * L_ + 2 * p) * C1 + c], a1 = s[(b * L_ + 2 * p + 1) * C1 + c];
  smx[(b * HP + p) * C1 + c] = fmaxf(a0, a1);
  float c0 = t[(b * L_ + 2 * p) * C1 + c], c1 = t[(b * L_ + 2 * p + 1) * C1 + c];
  tmx[(b * HP + p) * C1 + c] = fmaxf(c0, c1);
}

// ---------------------------------------------------------------------------
// K4: Conv2d(20->40,3x3) as implicit GEMM with WMMA.
// One block per image. The 48x192 f16 weight tile is DMA'd into LDS by the
// Tensor Data Mover (tensor_load_to_lds, TENSORcnt) while all 256 threads
// build the pooled plane P[c][25][25] = tm[pi][c]+sm[pj][c] in LDS.
// 8 waves sweep 34 M-tiles x 3 N-tiles, K=180 padded to 192 (6 WMMAs/tile).
// Per-block BN2 sum/sumsq go to per-block slots (deterministic).
// ---------------------------------------------------------------------------
__global__ void __launch_bounds__(256)
k4_conv2(const float* __restrict__ smx, const float* __restrict__ tmx,
         const _Float16* __restrict__ w2h, const float* __restrict__ b2,
         float* __restrict__ conv, float* __restrict__ part) {
  __shared__ _Float16 Pl[PLZ];        // 20 planes of 25x25 + zero pad region
  __shared__ _Float16 Bs[48 * K2P];   // [co][k] padded weights (TDM target)
  __shared__ int      koff[K2P];      // k -> plane offset (ci*625 + ky*25 + kx)
  __shared__ float    bsum[C2], bss[C2];
  const int b = blockIdx.x;
  const int tid = threadIdx.x;

  // Kick off the TDM copy of the weight tile (global f16 -> LDS), 2-D tile:
  // 48 rows x 192 cols x 2B, row stride 192 elements.
  if (tid == 0) {
    const unsigned long long ga = (unsigned long long)(uintptr_t)w2h;
    const unsigned int lds = (unsigned int)(uintptr_t)(&Bs[0]);
    u32x4 g0 = {};
    g0[0] = 1u;                                            // count=1, user mode
    g0[1] = lds;                                           // lds_addr
    g0[2] = (unsigned int)ga;                              // global_addr[31:0]
    g0[3] = (unsigned int)((ga >> 32) & 0x01FFFFFFu) | 0x80000000u; // addr[56:32] | type=2
    i32x8 g1 = {};
    g1[0] = (int)(1u << 16);      // data_size = 1 (2 bytes/elem)
    g1[1] = (int)(192u << 16);    // tensor_dim0 = 192 (bits 79:48, low half here)
    g1[2] = (int)(48u << 16);     // tensor_dim1 = 48  (bits 111:80, low half here)
    g1[3] = (int)(192u << 16);    // tile_dim0 = 192   (bits 127:112)
    g1[4] = (int)48;              // tile_dim1 = 48    (bits 143:128)
    g1[5] = (int)192;             // tensor_dim0_stride = 192 (bits 207:160 low)
    i32x4 z4 = {};
    i32x8 z8 = {};
    __builtin_amdgcn_tensor_load_to_lds(g0, g1, z4, z4, z8, 0);
  }

  // Meanwhile: build pooled plane + lookup table with all threads.
  for (int idx = tid; idx < PLZ; idx += 256) {
    float v = 0.f;
    if (idx < 12500) {
      int ci = idx / 625, rem = idx % 625;
      int pi = rem / 25, pj = rem % 25;
      v = tmx[(b * HP + pi) * C1 + ci] + smx[(b * HP + pj) * C1 + ci];
    }
    Pl[idx] = (_Float16)v;
  }
  for (int k = tid; k < K2P; k += 256) {
    int off = 12500;                                   // zeroed pad region
    if (k < K2) { int ci = k / 9, r9 = k % 9; off = ci * 625 + (r9 / 3) * 25 + (r9 % 3); }
    koff[k] = off;
  }
  if (tid < C2) { bsum[tid] = 0.f; bss[tid] = 0.f; }
  if (tid == 0) __builtin_amdgcn_s_wait_tensorcnt(0);  // weight tile landed
  __syncthreads();

  const int lane = tid & 31, wid = tid >> 5;
  const int m = lane & 15, h = lane >> 4;
  for (int job = wid; job < MT * 3; job += 8) {
    const int mt = job % MT, nt = job / MT;
    int pix = mt * 16 + m; if (pix >= NPIX) pix = NPIX - 1;   // clamp, rows not stored
    const int base = (pix / HC) * HP + (pix % HC);
    v8f acc = {};
    for (int k0 = 0; k0 < K2P; k0 += 32) {
      v16h a, bf;
#pragma unroll
      for (int e = 0; e < 8; ++e) a[e] = Pl[koff[k0 + 16 * h + e] + base];
#pragma unroll
      for (int e = 0; e < 8; ++e) a[8 + e] = Pl[koff[k0 + 16 * h + 8 + e] + base];
#pragma unroll
      for (int e = 0; e < 16; ++e) bf[e] = Bs[(nt * 16 + m) * K2P + k0 + 16 * h + e];
      acc = wmma_f16(a, bf, acc);
    }
    const int co = nt * 16 + m;
    if (co < C2) {
      const float bias = b2[co];
#pragma unroll
      for (int rr = 0; rr < 8; ++rr) {
        int M = mt * 16 + rr + 8 * h;
        if (M < NPIX) {
          float v = acc[rr] + bias;
          conv[((size_t)b * NPIX + M) * C2 + co] = v;
          atomicAdd(&bsum[co], v);
          atomicAdd(&bss[co], v * v);
        }
      }
    }
  }
  __syncthreads();
  if (tid < C2) {
    part[b * (2 * C2) + tid]      = bsum[tid];
    part[b * (2 * C2) + C2 + tid] = bss[tid];
  }
}

// K5: reduce per-block partials -> BN2 scale/shift
__global__ void k5_stats(const float* __restrict__ part, const float* __restrict__ g2,
                         const float* __restrict__ beta2, float* __restrict__ scsh) {
  const int c = threadIdx.x;
  if (c >= C2) return;
  float sum = 0.f, ss = 0.f;
  for (int b = 0; b < B_; ++b) { sum += part[b * 80 + c]; ss += part[b * 80 + 40 + c]; }
  const float invN = 1.f / ((float)B_ * NPIX);
  const float mu = sum * invN;
  const float var = ss * invN - mu * mu;
  const float sc = g2[c] * rsqrtf(var + 1e-5f);
  scsh[c] = sc;
  scsh[C2 + c] = beta2[c] - mu * sc;
}

// K6: BN2 + ReLU + 2x2 maxpool -> [B, 40*11*11] flat
__global__ void k6_bnpool(const float* __restrict__ conv, const float* __restrict__ scsh,
                          float* __restrict__ pool2) {
  const int b = blockIdx.x;
  for (int idx = threadIdx.x; idx < C2 * HO * HO; idx += blockDim.x) {
    const int c = idx / (HO * HO), rem = idx % (HO * HO);
    const int py = rem / HO, px = rem % HO;
    const float sc = scsh[c], sh = scsh[C2 + c];
    float mx = -1e30f;
#pragma unroll
    for (int dy = 0; dy < 2; ++dy)
#pragma unroll
      for (int dx = 0; dx < 2; ++dx) {
        float v = conv[((size_t)b * NPIX + (2 * py + dy) * HC + (2 * px + dx)) * C2 + c];
        mx = fmaxf(mx, fmaxf(0.f, v * sc + sh));
      }
    pool2[(size_t)b * 4840 + idx] = mx;
  }
}

// K7: FC1 — one wave per (b, j) dot product over K=4840
__global__ void __launch_bounds__(32)
k7_fc1(const float* __restrict__ pool2, const float* __restrict__ Wfc1,
       const float* __restrict__ bfc1, float* __restrict__ hbuf) {
  const int b = blockIdx.x, j = blockIdx.y, lane = threadIdx.x;
  const float* x = pool2 + (size_t)b * 4840;
  const float* w = Wfc1 + (size_t)j * 4840;
  float acc = 0.f;
  for (int k = lane; k < 4840; k += 32) acc += x[k] * w[k];
#pragma unroll
  for (int off = 16; off > 0; off >>= 1) acc += __shfl_down(acc, off, 32);
  if (lane == 0) hbuf[b * C1 + j] = acc + bfc1[j];
}

// K8: FC2 + sigmoid
__global__ void k8_fc2(const float* __restrict__ hbuf, const float* __restrict__ Wfc2,
                       const float* __restrict__ bfc2, float* __restrict__ out) {
  const int b = blockIdx.x * blockDim.x + threadIdx.x;
  if (b >= B_) return;
  float acc = bfc2[0];
#pragma unroll
  for (int j = 0; j < C1; ++j) acc += hbuf[b * C1 + j] * Wfc2[j];
  out[b] = 1.f / (1.f + __expf(-acc));
}

extern "C" void kernel_launch(void* const* d_in, const int* in_sizes, int n_in,
                              void* d_out, int out_size, void* d_ws, size_t ws_size,
                              hipStream_t stream) {
  const int*   src_tok = (const int*)d_in[0];
  const int*   trg_tok = (const int*)d_in[1];
  // d_in[2] = pad_idx (unused by the reference computation)
  const float* emb_src = (const float*)d_in[3];
  const float* emb_trg = (const float*)d_in[4];
  const float* W1   = (const float*)d_in[5];
  const float* b1   = (const float*)d_in[6];
  const float* g1   = (const float*)d_in[7];
  const float* be1  = (const float*)d_in[8];
  const float* W2   = (const float*)d_in[9];
  const float* b2   = (const float*)d_in[10];
  const float* g2   = (const float*)d_in[11];
  const float* be2  = (const float*)d_in[12];
  const float* Wfc1 = (const float*)d_in[13];
  const float* bfc1 = (const float*)d_in[14];
  const float* Wfc2 = (const float*)d_in[15];
  const float* bfc2 = (const float*)d_in[16];
  float* out = (float*)d_out;

  // Workspace layout (floats); total ~14.9M floats (~60 MB) + f16 weights
  float* ws    = (float*)d_ws;
  float* f_src = ws;                                  // 25600*20
  float* f_trg = f_src + 512000;                      // 25600*20
  float* smx   = f_trg + 512000;                      // 512*25*20
  float* tmx   = smx + 256000;                        // 512*25*20
  float* part  = tmx + 256000;                        // 512*80
  float* scsh  = part + 512 * 80;                     // 80
  float* conv  = scsh + 80;                           // 512*529*40
  float* pool2 = conv + (size_t)512 * 529 * 40;       // 512*4840
  float* hbuf  = pool2 + (size_t)512 * 4840;          // 512*20
  _Float16* w1h = (_Float16*)(hbuf + 512 * 20);       // 32*512 f16
  _Float16* w2h = w1h + 32 * 512;                     // 48*192 f16

  k0_prep<<<32, 256, 0, stream>>>(W1, W2, w1h, w2h);
  k1_embed_gemm<<<1600, 32, 0, stream>>>(src_tok, emb_src, w1h, b1, f_src);
  k1_embed_gemm<<<1600, 32, 0, stream>>>(trg_tok, emb_trg, w1h, b1, f_trg);
  k2_bn1<<<512, 32, 0, stream>>>(f_src, g1, be1);
  k2_bn1<<<512, 32, 0, stream>>>(f_trg, g1, be1);
  k3_poolmax<<<512, 512, 0, stream>>>(f_src, f_trg, smx, tmx);
  k4_conv2<<<512, 256, 0, stream>>>(smx, tmx, w2h, b2, conv, part);
  k5_stats<<<1, 64, 0, stream>>>(part, g2, be2, scsh);
  k6_bnpool<<<512, 256, 0, stream>>>(conv, scsh, pool2);
  k7_fc1<<<dim3(512, 20), 32, 0, stream>>>(pool2, Wfc1, bfc1, hbuf);
  k8_fc2<<<2, 256, 0, stream>>>(hbuf, Wfc2, bfc2, out);
}
